// GINNet_53815940219573
// MI455X (gfx1250) — compile-verified, matrix operand order checked by reading, not run on hardware
//
#include <hip/hip_runtime.h>
#include <hip/hip_bf16.h>

// ---------------- problem constants (match reference) ----------------
#define NN   50000
#define EE   400000
#define DD   512
#define HH   512
#define OO   256
#define BN_EPS 1e-5f

typedef __attribute__((ext_vector_type(16))) _Float16 v16h;
typedef __attribute__((ext_vector_type(8)))  float    v8f;

// ---------------- gather: xA[i,:] = emb[x_indices[i],:] ----------------
__global__ __launch_bounds__(256) void gather_k(const float* __restrict__ emb,
                                                const int* __restrict__ idx,
                                                float* __restrict__ xo) {
  int i = blockIdx.x * 256 + threadIdx.x;          // over N*512 = 25.6M
  if (i >= NN * DD) return;
  int row = i >> 9, col = i & 511;
  xo[i] = emb[((size_t)idx[row] << 9) + col];
}

// ---------------- scatter-add: agg[dst,:] += x[src,:] ----------------
// 128 lanes cover D/4=128 float4 chunks; 2 edges per 256-thread block.
__global__ __launch_bounds__(256) void scatter_add_k(const float4* __restrict__ x4,
                                                     const int* __restrict__ src,
                                                     const int* __restrict__ dst,
                                                     float* __restrict__ agg) {
  int e = blockIdx.x * 2 + (threadIdx.x >> 7);
  if (e >= EE) return;
  int f4 = threadIdx.x & 127;
  int s = src[e], d = dst[e];
  float4 v = x4[(size_t)s * 128 + f4];
  float* p = agg + (size_t)d * 512 + (size_t)f4 * 4;
  __hip_atomic_fetch_add(p + 0, v.x, __ATOMIC_RELAXED, __HIP_MEMORY_SCOPE_AGENT);
  __hip_atomic_fetch_add(p + 1, v.y, __ATOMIC_RELAXED, __HIP_MEMORY_SCOPE_AGENT);
  __hip_atomic_fetch_add(p + 2, v.z, __ATOMIC_RELAXED, __HIP_MEMORY_SCOPE_AGENT);
  __hip_atomic_fetch_add(p + 3, v.w, __ATOMIC_RELAXED, __HIP_MEMORY_SCOPE_AGENT);
}

// ---------------- fuse: h16 = f16(x + agg) ----------------
__global__ __launch_bounds__(256) void fuse_h16_k(const float* __restrict__ x,
                                                  const float* __restrict__ agg,
                                                  _Float16* __restrict__ h, int n) {
  int i = blockIdx.x * 256 + threadIdx.x;
  if (i < n) h[i] = (_Float16)(x[i] + agg[i]);
}

// ---------------- weight f32 [K,Nc] -> f16 transposed [Nc,K] ----------------
// Makes B-fragment loads contiguous along K (same pattern as A -> b128 loads).
__global__ __launch_bounds__(256) void cvt_t_h16_k(const float* __restrict__ w,
                                                   _Float16* __restrict__ wt,
                                                   int K, int Nc) {
  int i = blockIdx.x * 256 + threadIdx.x;          // over K*Nc (coalesced read)
  if (i >= K * Nc) return;
  int k = i / Nc, n = i - k * Nc;
  wt[(size_t)n * K + k] = (_Float16)w[i];
}

// ---------------- fragment loader ----------------
// 16-bit matrix fragment per ISA layout: lane r=lane&15 -> row, lane-half
// h=lane>>4 selects K sub-ranges {h*8..h*8+7, 16+h*8..16+h*8+7}. Row data is
// contiguous in memory, so this lowers to two global_load_b128.
__device__ __forceinline__ v16h load_frag(const _Float16* __restrict__ row,
                                          int k0, int h) {
  v16h f;
#pragma unroll
  for (int j = 0; j < 8; ++j) {
    f[j]     = row[k0 + h * 8 + j];
    f[8 + j] = row[k0 + 16 + h * 8 + j];
  }
  return f;
}

// ---------------- WMMA GEMM: out[M,Nc] = act(A[M,K] @ Wt^T + bias) ----------------
// A: [M,K] f16 row-major. Wt: [Nc,K] f16 row-major (pre-transposed weights).
// One wave32 per 16x64 output strip: 1 A-fragment feeds 4 WMMAs (4 N-tiles),
// K-loop of v_wmma_f32_16x16x32_f16 with f32 accumulation.
// C/D layout: lanes 0-15 N=lane M=v, lanes 16-31 N=lane-16 M=v+8.
template <bool RELU, bool OUT_F16>
__global__ __launch_bounds__(256) void wmma_gemm_k(const _Float16* __restrict__ A,
                                                   const _Float16* __restrict__ Wt,
                                                   const float* __restrict__ bias,
                                                   void* __restrict__ out,
                                                   int M, int K, int Nc) {
  const int lane = threadIdx.x & 31;
  const int wave = threadIdx.x >> 5;
  const int stripsN = Nc >> 6;                       // 64-wide output strips
  const long job = (long)blockIdx.x * 8 + wave;      // wave-uniform
  const long numJobs = (long)(M >> 4) * stripsN;
  if (job >= numJobs) return;                        // whole-wave exit: EXEC all-1s
  const int sn = (int)(job % stripsN);
  const int tm = (int)(job / stripsN);
  const int m0 = tm << 4, n0 = sn << 6;

  const int r = lane & 15;
  const int h = lane >> 4;

  v8f acc[4] = {v8f{}, v8f{}, v8f{}, v8f{}};
  const _Float16* __restrict__ arow = A + (size_t)(m0 + r) * K;
  const _Float16* __restrict__ brow[4] = {
      Wt + (size_t)(n0 + 0  + r) * K,
      Wt + (size_t)(n0 + 16 + r) * K,
      Wt + (size_t)(n0 + 32 + r) * K,
      Wt + (size_t)(n0 + 48 + r) * K,
  };

  for (int k0 = 0; k0 < K; k0 += 32) {
    const v16h a = load_frag(arow, k0, h);
#pragma unroll
    for (int t = 0; t < 4; ++t) {
      const v16h b = load_frag(brow[t], k0, h);
      acc[t] = __builtin_amdgcn_wmma_f32_16x16x32_f16(false, a, false, b,
                                                      (short)0, acc[t],
                                                      false, false);
    }
  }

  const int cm = m0 + ((lane >> 4) << 3);
#pragma unroll
  for (int t = 0; t < 4; ++t) {
    const int cn = n0 + 16 * t + (lane & 15);
    const float bv = bias[cn];
#pragma unroll
    for (int v = 0; v < 8; ++v) {
      float rr = acc[t][v] + bv;
      if (RELU) rr = fmaxf(rr, 0.0f);
      if (OUT_F16)
        ((_Float16*)out)[(size_t)(cm + v) * Nc + cn] = (_Float16)rr;
      else
        ((float*)out)[(size_t)(cm + v) * Nc + cn] = rr;
    }
  }
}

// ---------------- BN column stats: stats[c]=sum, stats[512+c]=sumsq ----------------
__global__ __launch_bounds__(256) void bn_stats_k(const float* __restrict__ y,
                                                  float* __restrict__ stats, int n) {
  const int col = blockIdx.x * 256 + threadIdx.x;     // gridDim.x = 2 (H=512)
  const int chunk = blockIdx.y, nch = gridDim.y;
  const int r0 = (int)(((long)n * chunk) / nch);
  const int r1 = (int)(((long)n * (chunk + 1)) / nch);
  float s = 0.0f, q = 0.0f;
  for (int rr = r0; rr < r1; ++rr) {
    float v = y[(size_t)rr * 512 + col];
    s += v;
    q += v * v;
  }
  __hip_atomic_fetch_add(&stats[col], s, __ATOMIC_RELAXED, __HIP_MEMORY_SCOPE_AGENT);
  __hip_atomic_fetch_add(&stats[512 + col], q, __ATOMIC_RELAXED, __HIP_MEMORY_SCOPE_AGENT);
}

// ---------------- BN apply + ReLU: xo = relu((y-mu)*rsqrt(var+eps)*g + be) ----------------
__global__ __launch_bounds__(256) void bn_apply_k(const float* __restrict__ y,
                                                  const float* __restrict__ stats,
                                                  const float* __restrict__ g,
                                                  const float* __restrict__ be,
                                                  float* __restrict__ xo, int nrows) {
  int i = blockIdx.x * 256 + threadIdx.x;
  if (i >= nrows * 512) return;
  int col = i & 511;
  float invn = 1.0f / (float)nrows;
  float mu = stats[col] * invn;
  float var = stats[512 + col] * invn - mu * mu;
  float sc = rsqrtf(var + BN_EPS) * g[col];
  float v = (y[i] - mu) * sc + be[col];
  xo[i] = fmaxf(v, 0.0f);
}

// ---------------- driver ----------------
extern "C" void kernel_launch(void* const* d_in, const int* in_sizes, int n_in,
                              void* d_out, int out_size, void* d_ws, size_t ws_size,
                              hipStream_t stream) {
  (void)in_sizes; (void)n_in; (void)out_size; (void)ws_size;

  const int*   x_indices = (const int*)d_in[0];
  const int*   ei        = (const int*)d_in[1];     // [2,E]: src=ei[0..E), dst=ei[E..2E)
  const float* emb       = (const float*)d_in[2];
  const float* w1[3] = {(const float*)d_in[3],  (const float*)d_in[9],  (const float*)d_in[15]};
  const float* b1[3] = {(const float*)d_in[4],  (const float*)d_in[10], (const float*)d_in[16]};
  const float* w2[3] = {(const float*)d_in[5],  (const float*)d_in[11], (const float*)d_in[17]};
  const float* b2[3] = {(const float*)d_in[6],  (const float*)d_in[12], (const float*)d_in[18]};
  const float* gg[2] = {(const float*)d_in[7],  (const float*)d_in[13]};
  const float* bb[2] = {(const float*)d_in[8],  (const float*)d_in[14]};

  // workspace layout (~308 MB):
  //   xA  : N*512 f32  (current features)
  //   xB  : N*512 f32  (agg during aggregate phase, GEMM2 output afterwards - lifetimes disjoint)
  //   h16 : N*512 f16  (GEMM1 input)
  //   m16 : N*512 f16  (GEMM1 output / GEMM2 input)
  //   w1t : 512*512 f16 (transposed), w2t : 512*512 f16 (transposed), stats : 1024 f32
  const size_t NF = (size_t)NN * 512;
  float*    xA  = (float*)d_ws;
  float*    xB  = xA + NF;
  _Float16* h16 = (_Float16*)(xB + NF);
  _Float16* m16 = h16 + NF;
  _Float16* w1t = m16 + NF;
  _Float16* w2t = w1t + (size_t)512 * 512;
  float*    stats = (float*)(w2t + (size_t)512 * 512);

  const int nElem = NN * 512;                 // 25.6M
  const int eb256 = (nElem + 255) / 256;

  // x = emb[x_indices]
  gather_k<<<eb256, 256, 0, stream>>>(emb, x_indices, xA);

  for (int l = 0; l < 3; ++l) {
    const int Nc2 = (l == 2) ? OO : HH;       // second-matmul output width

    // agg = segment_sum(x[src], dst)
    hipMemsetAsync(xB, 0, NF * sizeof(float), stream);
    scatter_add_k<<<EE / 2, 256, 0, stream>>>((const float4*)xA, ei, ei + EE, xB);

    // h16 = f16(x + agg)
    fuse_h16_k<<<eb256, 256, 0, stream>>>(xA, xB, h16, nElem);

    // weights -> f16, transposed to [Nc,K]
    cvt_t_h16_k<<<(512 * 512 + 255) / 256, 256, 0, stream>>>(w1[l], w1t, 512, 512);
    cvt_t_h16_k<<<(512 * Nc2 + 255) / 256, 256, 0, stream>>>(w2[l], w2t, 512, Nc2);

    // mid = relu(h @ w1 + b1)   [N,512] f16
    {
      long jobs = (long)(NN / 16) * (512 / 64);
      int blocks = (int)((jobs + 7) / 8);
      wmma_gemm_k<true, true><<<blocks, 256, 0, stream>>>(h16, w1t, b1[l], m16,
                                                          NN, 512, 512);
    }

    if (l < 2) {
      // y = mid @ w2 + b2   [N,512] f32 -> xB
      long jobs = (long)(NN / 16) * (512 / 64);
      int blocks = (int)((jobs + 7) / 8);
      wmma_gemm_k<false, false><<<blocks, 256, 0, stream>>>(m16, w2t, b2[l], xB,
                                                            NN, 512, 512);
      // BN (batch stats) + ReLU -> xA
      hipMemsetAsync(stats, 0, 1024 * sizeof(float), stream);
      bn_stats_k<<<dim3(2, 512), 256, 0, stream>>>(xB, stats, NN);
      bn_apply_k<<<eb256, 256, 0, stream>>>(xB, stats, gg[l], bb[l], xA, NN);
    } else {
      // out = mid @ w2 + b2   [N,256] f32 -> d_out
      long jobs = (long)(NN / 16) * (OO / 64);
      int blocks = (int)((jobs + 7) / 8);
      wmma_gemm_k<false, false><<<blocks, 256, 0, stream>>>(m16, w2t, b2[l],
                                                            (float*)d_out,
                                                            NN, 512, OO);
    }
  }
}
